// SelfAttention_44358422233369
// MI455X (gfx1250) — compile-verified, matrix operand order checked by reading, not run on hardware
//
#include <hip/hip_runtime.h>
#include <hip/hip_bf16.h>

#define SEQ  2048
#define BATCH 4
#define EMB  1024

typedef __bf16 bf16;
typedef __attribute__((ext_vector_type(16))) __bf16 v16bf;
typedef __attribute__((ext_vector_type(8)))  __bf16 v8bf;
typedef __attribute__((ext_vector_type(8)))  float  v8f;

// Assemble a 16x32 bf16 fragment from two 16-byte chunks at p and p+16 elems.
// CDNA5 16-bit A/B layout: lanes 0-15 hold K 0..7 & 16..23, lanes 16-31 hold
// K 8..15 & 24..31 (kh = (lane>>4)*8 handled by the caller's pointer).
__device__ __forceinline__ v16bf frag_from(const bf16* p) {
  v8bf lo = *(const v8bf*)(p);
  v8bf hi = *(const v8bf*)(p + 16);
  v16bf r;
#pragma unroll
  for (int i = 0; i < 8; ++i) { r[i] = lo[i]; r[i + 8] = hi[i]; }
  return r;
}

__device__ __forceinline__ v16bf load_frag(const bf16* __restrict__ base, int ld) {
  const int lane = threadIdx.x & 31;
  return frag_from(base + (size_t)(lane & 15) * ld + (((lane >> 4)) << 3));
}

__device__ __forceinline__ v8f wmma_bf16(v16bf a, v16bf b, v8f c) {
  return __builtin_amdgcn_wmma_f32_16x16x32_bf16(false, a, false, b, (short)0, c,
                                                 false, false);
}

__global__ void cvt_f32_to_bf16(const float* __restrict__ src,
                                bf16* __restrict__ dst, int n) {
  int i = blockIdx.x * blockDim.x + threadIdx.x;
  if (i < n) dst[i] = (bf16)src[i];
}

// ---------------------------------------------------------------------------
// QKV projection: C[m,n] = sum_k A[m,k]*W[n,k] + bias[n]
// Block tile 128x128 (8 waves, each 32x64).  W slab (128 rows x 32 k) staged
// in LDS via CDNA5 async-to-LDS DMA, double buffered (ASYNCcnt + barriers).
// transposed==0: out bf16 [M,N]; transposed==1: out bf16 [B][N][SEQ] (V^T).
// ---------------------------------------------------------------------------
#define LROW 40  // bf16 elements per LDS row (80B: conflict-free b128 reads)

__global__ void __launch_bounds__(256) gemm_qkv(
    const bf16* __restrict__ A, const bf16* __restrict__ W,
    const float* __restrict__ bias, bf16* __restrict__ out,
    int N, int K, int transposed) {
  __shared__ __attribute__((aligned(16))) bf16 smem[2][128 * LROW];

  const int tid  = threadIdx.x;
  const int w    = tid >> 5;
  const int lane = tid & 31;
  const int n    = lane & 15;
  const int mh   = lane >> 4;
  const int kh   = mh << 3;
  const int mw   = w >> 1;                       // 0..3
  const int nw   = w & 1;                        // 0..1
  const int m0   = blockIdx.x * 128 + mw * 32;
  const int nblk = blockIdx.y * 128;
  const int n0   = nblk + nw * 64;

  // Cooperative copy: 128 rows x 64B; each thread moves 32B (one b128 with
  // inst_offset 0/16 applied to both LDS and global address).
  const int      crow  = tid >> 1;
  const int      chalf = tid & 1;
  const uint64_t gsrc  = (uint64_t)(W + (size_t)(nblk + crow) * K + chalf * 16);
  // smem is the only __shared__ object -> LDS byte offset 0.  The smem pointer
  // is passed to the asm (unused in the template) so it ESCAPES: the "memory"
  // clobber then forces LLVM to treat LDS as written by the DMA, keeping the
  // ds_load fragment reads alive.
  void* smem_escape = (void*)smem;
  const unsigned lbase = (unsigned)(crow * LROW + chalf * 16) * 2;
  const unsigned lsz   = 128 * LROW * 2;

  v8f acc[8] = {};
  {
    asm volatile("global_load_async_to_lds_b128 %0, %1, off"
                 :: "v"(lbase), "v"(gsrc), "r"(smem_escape) : "memory");
    asm volatile("global_load_async_to_lds_b128 %0, %1, off offset:16"
                 :: "v"(lbase), "v"(gsrc), "r"(smem_escape) : "memory");
  }
  v16bf a0 = load_frag(A + (size_t)m0 * K, K);
  v16bf a1 = load_frag(A + (size_t)(m0 + 16) * K, K);

  const int NK = K / 32;
  for (int kk = 0; kk < NK; ++kk) {
    const int buf = kk & 1;
    if (kk + 1 < NK) {
      const uint64_t g    = gsrc + (uint64_t)(kk + 1) * 64;
      const unsigned ldst = (buf ^ 1) ? (lbase + lsz) : lbase;
      asm volatile("global_load_async_to_lds_b128 %0, %1, off"
                   :: "v"(ldst), "v"(g), "r"(smem_escape) : "memory");
      asm volatile("global_load_async_to_lds_b128 %0, %1, off offset:16"
                   :: "v"(ldst), "v"(g), "r"(smem_escape) : "memory");
      asm volatile("s_wait_asynccnt 0x2" ::: "memory");
    } else {
      asm volatile("s_wait_asynccnt 0x0" ::: "memory");
    }
    __syncthreads();

    v16bf a0n = a0, a1n = a1;
    if (kk + 1 < NK) {  // prefetch next A fragments while WMMAs run
      a0n = load_frag(A + (size_t)m0 * K + (kk + 1) * 32, K);
      a1n = load_frag(A + (size_t)(m0 + 16) * K + (kk + 1) * 32, K);
    }
#pragma unroll
    for (int j = 0; j < 4; ++j) {
      v16bf bfj = frag_from(&smem[buf][(nw * 64 + j * 16 + n) * LROW + kh]);
      acc[j]     = wmma_bf16(a0, bfj, acc[j]);
      acc[4 + j] = wmma_bf16(a1, bfj, acc[4 + j]);
    }
    a0 = a0n; a1 = a1n;
    __syncthreads();
  }

#pragma unroll
  for (int j = 0; j < 4; ++j) {
    float bvs = bias[n0 + j * 16 + n];
#pragma unroll
    for (int r = 0; r < 8; ++r) { acc[j][r] += bvs; acc[4 + j][r] += bvs; }
  }

  if (!transposed) {
#pragma unroll
    for (int jm = 0; jm < 2; ++jm)
#pragma unroll
      for (int j = 0; j < 4; ++j) {
        bf16* p = out + (size_t)(m0 + jm * 16 + mh * 8) * N + n0 + j * 16 + n;
#pragma unroll
        for (int r = 0; r < 8; ++r) p[(size_t)r * N] = (bf16)acc[jm * 4 + j][r];
      }
  } else {
    const int b = m0 / SEQ;
#pragma unroll
    for (int jm = 0; jm < 2; ++jm) {
      const int s = (m0 % SEQ) + jm * 16 + mh * 8;
#pragma unroll
      for (int j = 0; j < 4; ++j) {
        v8bf v;
#pragma unroll
        for (int r = 0; r < 8; ++r) v[r] = (bf16)acc[jm * 4 + j][r];
        *(v8bf*)(out + ((size_t)(b * N + n0 + j * 16 + n)) * SEQ + s) = v;
      }
    }
  }
}

// ---------------------------------------------------------------------------
// Attention scores + softmax -> normalized bf16 P (two WMMA passes, online
// stats in registers, K fragments register-pipelined).
// ---------------------------------------------------------------------------
__device__ __forceinline__ v8f qk_tile(const v16bf (&qf)[32],
                                       const bf16* __restrict__ kbase) {
  v8f c = {};
  v16bf kf = load_frag(kbase, EMB);
#pragma unroll
  for (int e = 0; e < 32; ++e) {
    v16bf kn = kf;
    if (e + 1 < 32) kn = load_frag(kbase + (e + 1) * 32, EMB);
    c = wmma_bf16(qf[e], kf, c);
    kf = kn;
  }
  return c;
}

__global__ void __launch_bounds__(256) attn_softmax_p(
    const bf16* __restrict__ Q,   // [B*SEQ, EMB]
    const bf16* __restrict__ Km,  // [B*SEQ, EMB]
    bf16* __restrict__ P) {       // [B, SEQ, SEQ]
  const int w    = threadIdx.x >> 5;
  const int lane = threadIdx.x & 31;
  const int n    = lane & 15;
  const int mh   = lane >> 4;
  const int tile = blockIdx.x * 8 + w;
  const int b    = tile >> 7;
  const int qt   = tile & 127;
  const int q0   = qt * 16;
  const float scale = 0.03125f;  // 1/sqrt(1024)

  v16bf qf[32];
  const bf16* qbase = Q + ((size_t)b * SEQ + q0) * EMB;
#pragma unroll
  for (int e = 0; e < 32; ++e) qf[e] = load_frag(qbase + e * 32, EMB);

  float mrow[8], lrow[8];
#pragma unroll
  for (int r = 0; r < 8; ++r) { mrow[r] = -3.0e38f; lrow[r] = 0.0f; }

  for (int kt = 0; kt <= qt; ++kt) {  // pass A: stats
    v8f c = qk_tile(qf, Km + ((size_t)b * SEQ + kt * 16) * EMB);
    const int kg = kt * 16 + n;
#pragma unroll
    for (int r = 0; r < 8; ++r) {
      const int qg = q0 + mh * 8 + r;
      float s = (kg <= qg) ? c[r] * scale : -3.0e38f;
      float t = s;
#pragma unroll
      for (int off = 8; off >= 1; off >>= 1) t = fmaxf(t, __shfl_xor(t, off, 32));
      float mnew = fmaxf(mrow[r], t);
      float su = __expf(s - mnew);
#pragma unroll
      for (int off = 8; off >= 1; off >>= 1) su += __shfl_xor(su, off, 32);
      lrow[r] = lrow[r] * __expf(mrow[r] - mnew) + su;
      mrow[r] = mnew;
    }
  }

  float rinv[8];
#pragma unroll
  for (int r = 0; r < 8; ++r) rinv[r] = 1.0f / lrow[r];

  const int ktpad = (qt + 2) & ~1;  // zero-pad key span to k % 32 == 0
  for (int kt = 0; kt < ktpad; ++kt) {  // pass B: write P
    bf16* pb = P + (size_t)b * SEQ * SEQ + (size_t)q0 * SEQ + kt * 16 + n;
    if (kt <= qt) {
      v8f c = qk_tile(qf, Km + ((size_t)b * SEQ + kt * 16) * EMB);
      const int kg = kt * 16 + n;
#pragma unroll
      for (int r = 0; r < 8; ++r) {
        const int qg = q0 + mh * 8 + r;
        float p = (kg <= qg) ? __expf(c[r] * scale - mrow[r]) * rinv[r] : 0.0f;
        pb[(size_t)(mh * 8 + r) * SEQ] = (bf16)p;
      }
    } else {
#pragma unroll
      for (int r = 0; r < 8; ++r) pb[(size_t)(mh * 8 + r) * SEQ] = (bf16)0.0f;
    }
  }
}

// ---------------------------------------------------------------------------
// O[b,q,e] = sum_k P[b,q,k] * Vt[b,e,k], causal bound per 32-row wave tile.
// No barriers (divergent trip counts); register-pipelined fragments.
// ---------------------------------------------------------------------------
__global__ void __launch_bounds__(256) gemm_pv(
    const bf16* __restrict__ P,   // [B,SEQ,SEQ]
    const bf16* __restrict__ Vt,  // [B,EMB,SEQ]
    float* __restrict__ O) {      // [B,SEQ,EMB]
  const int w    = threadIdx.x >> 5;
  const int lane = threadIdx.x & 31;
  const int n    = lane & 15;
  const int mh   = lane >> 4;
  const int mw   = w >> 1;
  const int nw   = w & 1;
  const int b    = blockIdx.x >> 4;
  const int q0   = (blockIdx.x & 15) * 128 + mw * 32;
  const int n0   = blockIdx.y * 128 + nw * 64;

  const bf16* Pq = P + (size_t)b * SEQ * SEQ + (size_t)q0 * SEQ;
  const bf16* Vb = Vt + (size_t)b * EMB * SEQ;

  v8f acc[8] = {};
  v16bf a0 = load_frag(Pq, SEQ);
  v16bf a1 = load_frag(Pq + (size_t)16 * SEQ, SEQ);
  v16bf bv[4];
#pragma unroll
  for (int j = 0; j < 4; ++j) bv[j] = load_frag(Vb + (size_t)(n0 + j * 16) * SEQ, SEQ);

  const int NK = q0 / 32 + 1;  // P is defined (value or 0) for k < q0+32
  for (int kk = 0; kk < NK; ++kk) {
    v16bf a0n = a0, a1n = a1, bvn[4] = {bv[0], bv[1], bv[2], bv[3]};
    if (kk + 1 < NK) {
      const int k0 = (kk + 1) * 32;
      a0n = load_frag(Pq + k0, SEQ);
      a1n = load_frag(Pq + (size_t)16 * SEQ + k0, SEQ);
#pragma unroll
      for (int j = 0; j < 4; ++j)
        bvn[j] = load_frag(Vb + (size_t)(n0 + j * 16) * SEQ + k0, SEQ);
    }
#pragma unroll
    for (int j = 0; j < 4; ++j) {
      acc[j]     = wmma_bf16(a0, bv[j], acc[j]);
      acc[4 + j] = wmma_bf16(a1, bv[j], acc[4 + j]);
    }
    a0 = a0n; a1 = a1n;
#pragma unroll
    for (int j = 0; j < 4; ++j) bv[j] = bvn[j];
  }

#pragma unroll
  for (int jm = 0; jm < 2; ++jm) {
    float* Ob = O + ((size_t)b * SEQ + q0 + jm * 16 + mh * 8) * EMB + n0 + n;
#pragma unroll
    for (int j = 0; j < 4; ++j)
#pragma unroll
      for (int r = 0; r < 8; ++r) Ob[(size_t)r * EMB + j * 16] = acc[jm * 4 + j][r];
  }
}

extern "C" void kernel_launch(void* const* d_in, const int* in_sizes, int n_in,
                              void* d_out, int out_size, void* d_ws, size_t ws_size,
                              hipStream_t stream) {
  const float* x  = (const float*)d_in[0];
  const float* Wq = (const float*)d_in[1];
  const float* bq = (const float*)d_in[2];
  const float* Wk = (const float*)d_in[3];
  const float* bk = (const float*)d_in[4];
  const float* Wv = (const float*)d_in[5];
  const float* bv = (const float*)d_in[6];

  const size_t ME = (size_t)BATCH * SEQ * EMB;
  const size_t WE = (size_t)EMB * EMB;

  bf16* ws  = (bf16*)d_ws;
  bf16* xbf = ws;  ws += ME;
  bf16* wqb = ws;  ws += WE;
  bf16* wkb = ws;  ws += WE;
  bf16* wvb = ws;  ws += WE;
  bf16* qbf = ws;  ws += ME;
  bf16* kbf = ws;  ws += ME;
  bf16* vtb = ws;  ws += ME;
  bf16* pbf = ws;  // [B,SEQ,SEQ]

  cvt_f32_to_bf16<<<(int)(ME / 256), 256, 0, stream>>>(x,  xbf, (int)ME);
  cvt_f32_to_bf16<<<(int)(WE / 256), 256, 0, stream>>>(Wq, wqb, (int)WE);
  cvt_f32_to_bf16<<<(int)(WE / 256), 256, 0, stream>>>(Wk, wkb, (int)WE);
  cvt_f32_to_bf16<<<(int)(WE / 256), 256, 0, stream>>>(Wv, wvb, (int)WE);

  dim3 g1(BATCH * SEQ / 128, EMB / 128);
  gemm_qkv<<<g1, 256, 0, stream>>>(xbf, wqb, bq, qbf, EMB, EMB, 0);
  gemm_qkv<<<g1, 256, 0, stream>>>(xbf, wkb, bk, kbf, EMB, EMB, 0);
  gemm_qkv<<<g1, 256, 0, stream>>>(xbf, wvb, bv, vtb, EMB, EMB, 1);

  attn_softmax_p<<<BATCH * (SEQ / 16) / 8, 256, 0, stream>>>(qbf, kbf, pbf);

  dim3 g2(BATCH * (SEQ / 128), EMB / 128);
  gemm_pv<<<g2, 256, 0, stream>>>(pbf, vtb, (float*)d_out);
}
